// tokenchange_1537598292235
// MI455X (gfx1250) — matrix operand order repaired and checked by reading, not compiled, not run on hardware
//
#include <hip/hip_runtime.h>

#define NB 8
#define LL 2048
#define DD 128
#define NLROWS (NB * LL)      // 16384
#define NLD (NB * LL * DD)    // 2097152

typedef __attribute__((ext_vector_type(16))) __bf16 v16bf;
typedef __attribute__((ext_vector_type(8)))  float  v8f;

union BF16x16 {
    v16bf v;
    unsigned short s[16];
    uint4 q[2];
};

// f32 -> bf16 round-to-nearest-even
__device__ __forceinline__ unsigned short f2bf(float f) {
    union { float f; unsigned int u; } x; x.f = f;
    unsigned int u = x.u;
    u = u + 0x7FFFu + ((u >> 16) & 1u);
    return (unsigned short)(u >> 16);
}

// contiguous 16 bf16 (32B) -> B fragment
__device__ __forceinline__ v16bf load_bf16x16(const unsigned short* p) {
    BF16x16 u;
    u.q[0] = *(const uint4*)(p);
    u.q[1] = *(const uint4*)(p + 8);
    return u.v;
}

// two contiguous 8-bf16 chunks -> A fragment
__device__ __forceinline__ v16bf load_bf8x2(const unsigned short* p0, const unsigned short* p1) {
    BF16x16 u;
    u.q[0] = *(const uint4*)(p0);
    u.q[1] = *(const uint4*)(p1);
    return u.v;
}

// monotone float->uint key
__device__ __forceinline__ unsigned int fkey(float f) {
    union { float f; unsigned int u; } x; x.f = f;
    unsigned int u = x.u;
    return (u & 0x80000000u) ? ~u : (u | 0x80000000u);
}

__global__ void zero_f32_kernel(float* p, int n) {
    int i = blockIdx.x * blockDim.x + threadIdx.x;
    if (i < n) p[i] = 0.0f;
}

__global__ void convert_w_kernel(const float* __restrict__ W, unsigned short* __restrict__ Wb) {
    int i = blockIdx.x * blockDim.x + threadIdx.x;
    if (i < DD * DD) Wb[i] = f2bf(W[i]);
}

// pt = x @ W^T + b (stored bf16). grid: (NLROWS/16, 2), block 128 (4 waves).
__global__ void proj_kernel(const float* __restrict__ x_t, const float* __restrict__ x_f,
                            const unsigned short* __restrict__ Wb, const float* __restrict__ bias,
                            unsigned short* __restrict__ pt, unsigned short* __restrict__ pf) {
    const int tid  = threadIdx.x;
    const int wave = tid >> 5;
    const int lane = tid & 31;
    const int half = lane >> 4;
    const int r16  = lane & 15;
    const int l0   = blockIdx.x * 16;

    const float* x     = (blockIdx.y == 0) ? x_t : x_f;
    unsigned short* po = (blockIdx.y == 0) ? pt : pf;

    const float* xr = x + (size_t)(l0 + r16) * DD;

    v16bf a[4];
#pragma unroll
    for (int ks = 0; ks < 4; ++ks) {
        const int k0 = ks * 32;
        const int o1 = k0 + (half ? 16 : 0);
        const int o2 = k0 + (half ? 24 : 8);
        BF16x16 u;
#pragma unroll
        for (int j = 0; j < 8; ++j) u.s[j] = f2bf(xr[o1 + j]);
#pragma unroll
        for (int j = 0; j < 8; ++j) u.s[8 + j] = f2bf(xr[o2 + j]);
        a[ks] = u.v;
    }

#pragma unroll
    for (int t = 0; t < 2; ++t) {
        const int d0   = (wave * 2 + t) * 16;
        const int dcol = d0 + r16;
        const unsigned short* wr = Wb + (size_t)dcol * DD + (half ? 16 : 0);
        v8f acc = {};
#pragma unroll
        for (int ks = 0; ks < 4; ++ks) {
            v16bf bfrag = load_bf16x16(wr + ks * 32);
            acc = __builtin_amdgcn_wmma_f32_16x16x32_bf16(false, a[ks], false, bfrag,
                                                          (short)0, acc, false, false);
        }
        const float bb = bias[dcol];
#pragma unroll
        for (int r = 0; r < 8; ++r) {
            const int rowg = l0 + r + half * 8;
            po[(size_t)rowg * DD + dcol] = f2bf(acc[r] + bb);
        }
    }
}

// score[type][n][m] = sum_{l != m} softmax_row_l(P P^T * scale)[m]
// grid: (LL/32, NB, 2), block 128 (4 waves). 32-row strip (two A-tiles) per block:
// each loaded B fragment feeds 2 WMMAs (32 FLOP/byte from L2).
// Pass 1: per-lane online softmax (no shuffles in tile loop; exp co-executes with WMMA),
// hierarchical merge (lane -> wave via LDS). Pass 2: p = exp(s - (M + ln Z)), diag zeroed,
// column sums in LDS (disjoint per wave), one f32 atomicAdd per column per block.
__global__ void score_kernel(const unsigned short* __restrict__ pt,
                             const unsigned short* __restrict__ pf,
                             float* __restrict__ score) {
    const int tid  = threadIdx.x;
    const int wave = tid >> 5;
    const int lane = tid & 31;
    const int half = lane >> 4;
    const int r16  = lane & 15;
    const int l0   = blockIdx.x * 32;
    const int n    = blockIdx.y;
    const int type = blockIdx.z;

    const float scale = 0.08838834764831845f; // 1/sqrt(128)
    const unsigned short* M = type ? pf : pt;
    const unsigned short* base = M + (size_t)n * LL * DD;

    __shared__ float lm[4][32];
    __shared__ float lz[4][32];
    __shared__ float lds_col[LL];

    // Two A-tiles (rows l0..l0+15 and l0+16..l0+31), resident for the whole kernel
    const unsigned short* ar0 = base + (size_t)(l0 + r16) * DD;
    const unsigned short* ar1 = base + (size_t)(l0 + 16 + r16) * DD;
    v16bf a0[4], a1[4];
#pragma unroll
    for (int ks = 0; ks < 4; ++ks) {
        const int k0 = ks * 32;
        const int o1 = k0 + (half ? 16 : 0);
        const int o2 = k0 + (half ? 24 : 8);
        a0[ks] = load_bf8x2(ar0 + o1, ar0 + o2);
        a1[ks] = load_bf8x2(ar1 + o1, ar1 + o2);
    }

    // -------- pass 1: per-lane online softmax over this lane's column stream --------
    float m0v[8], z0v[8], m1v[8], z1v[8];
#pragma unroll
    for (int r = 0; r < 8; ++r) {
        m0v[r] = -3.0e38f; z0v[r] = 0.0f;
        m1v[r] = -3.0e38f; z1v[r] = 0.0f;
    }

    for (int mt = wave; mt < LL / 16; mt += 4) {
        const int m0 = mt * 16;
        const unsigned short* cr = base + (size_t)(m0 + r16) * DD + (half ? 16 : 0);
        v8f acc0 = {}, acc1 = {};
#pragma unroll
        for (int ks = 0; ks < 4; ++ks) {
            v16bf bfrag = load_bf16x16(cr + ks * 32);
            acc0 = __builtin_amdgcn_wmma_f32_16x16x32_bf16(false, a0[ks], false, bfrag,
                                                           (short)0, acc0, false, false);
            acc1 = __builtin_amdgcn_wmma_f32_16x16x32_bf16(false, a1[ks], false, bfrag,
                                                           (short)0, acc1, false, false);
        }
#pragma unroll
        for (int r = 0; r < 8; ++r) {
            {
                const float v  = acc0[r] * scale;
                const float nm = fmaxf(m0v[r], v);
                z0v[r] = z0v[r] * __expf(m0v[r] - nm) + __expf(v - nm);
                m0v[r] = nm;
            }
            {
                const float v  = acc1[r] * scale;
                const float nm = fmaxf(m1v[r], v);
                z1v[r] = z1v[r] * __expf(m1v[r] - nm) + __expf(v - nm);
                m1v[r] = nm;
            }
        }
    }

    // merge (m,z) across the 16 lanes of each half
#pragma unroll
    for (int r = 0; r < 8; ++r) {
#pragma unroll
        for (int off = 1; off < 16; off <<= 1) {
            {
                const float om = __shfl_xor(m0v[r], off);
                const float oz = __shfl_xor(z0v[r], off);
                const float nm = fmaxf(m0v[r], om);
                z0v[r] = z0v[r] * __expf(m0v[r] - nm) + oz * __expf(om - nm);
                m0v[r] = nm;
            }
            {
                const float om = __shfl_xor(m1v[r], off);
                const float oz = __shfl_xor(z1v[r], off);
                const float nm = fmaxf(m1v[r], om);
                z1v[r] = z1v[r] * __expf(m1v[r] - nm) + oz * __expf(om - nm);
                m1v[r] = nm;
            }
        }
    }

    // publish per-wave stats (strip-row index: tile0 = r+8*half, tile1 = 16+r+8*half)
    if (lane == 0) {
#pragma unroll
        for (int r = 0; r < 8; ++r) {
            lm[wave][r] = m0v[r];       lz[wave][r] = z0v[r];
            lm[wave][16 + r] = m1v[r];  lz[wave][16 + r] = z1v[r];
        }
    }
    if (lane == 16) {
#pragma unroll
        for (int r = 0; r < 8; ++r) {
            lm[wave][8 + r] = m0v[r];   lz[wave][8 + r] = z0v[r];
            lm[wave][24 + r] = m1v[r];  lz[wave][24 + r] = z1v[r];
        }
    }
    for (int i = tid; i < LL; i += 128) lds_col[i] = 0.0f;
    __syncthreads();

    // cross-wave merge; fold 1/Z into exponent: Mlog = M + ln(Z)
    float Ml0[8], Ml1[8];
#pragma unroll
    for (int r = 0; r < 8; ++r) {
        {
            const int rg = half * 8 + r;
            float mm = fmaxf(fmaxf(lm[0][rg], lm[1][rg]), fmaxf(lm[2][rg], lm[3][rg]));
            float zz = 0.0f;
#pragma unroll
            for (int w = 0; w < 4; ++w) zz += lz[w][rg] * __expf(lm[w][rg] - mm);
            Ml0[r] = mm + __logf(zz);
        }
        {
            const int rg = 16 + half * 8 + r;
            float mm = fmaxf(fmaxf(lm[0][rg], lm[1][rg]), fmaxf(lm[2][rg], lm[3][rg]));
            float zz = 0.0f;
#pragma unroll
            for (int w = 0; w < 4; ++w) zz += lz[w][rg] * __expf(lm[w][rg] - mm);
            Ml1[r] = mm + __logf(zz);
        }
    }

    // -------- pass 2: recompute tiles, normalize, column-sum --------
    for (int mt = wave; mt < LL / 16; mt += 4) {
        const int m0 = mt * 16;
        const unsigned short* cr = base + (size_t)(m0 + r16) * DD + (half ? 16 : 0);
        v8f acc0 = {}, acc1 = {};
#pragma unroll
        for (int ks = 0; ks < 4; ++ks) {
            v16bf bfrag = load_bf16x16(cr + ks * 32);
            acc0 = __builtin_amdgcn_wmma_f32_16x16x32_bf16(false, a0[ks], false, bfrag,
                                                           (short)0, acc0, false, false);
            acc1 = __builtin_amdgcn_wmma_f32_16x16x32_bf16(false, a1[ks], false, bfrag,
                                                           (short)0, acc1, false, false);
        }
        const int col = m0 + r16;
        float colpart = 0.0f;
#pragma unroll
        for (int r = 0; r < 8; ++r) {
            const int rowa = l0 + r + half * 8;
            const int rowb = rowa + 16;
            float p0 = __expf(acc0[r] * scale - Ml0[r]);
            float p1 = __expf(acc1[r] * scale - Ml1[r]);
            if (rowa == col) p0 = 0.0f; // diagonal zeroed AFTER softmax
            if (rowb == col) p1 = 0.0f;
            colpart += p0 + p1;
        }
        colpart += __shfl_xor(colpart, 16);
        if (half == 0) lds_col[col] += colpart; // waves own disjoint col tiles
    }
    __syncthreads();

    float* srow = score + (size_t)(type * NB + n) * LL;
    for (int i = tid; i < LL; i += 128) atomicAdd(&srow[i], lds_col[i]);
}

// per (type,n) row: theta = 1024-th smallest; mask = score <= theta.
__global__ void mask_kernel(const float* __restrict__ score, unsigned char* __restrict__ masks) {
    const int tid = threadIdx.x;
    const float* srow = score + (size_t)blockIdx.x * LL;
    unsigned char* mrow = masks + (size_t)blockIdx.x * LL;

    unsigned int keys[8];
#pragma unroll
    for (int j = 0; j < 8; ++j) keys[j] = fkey(srow[tid + j * 256]);

    __shared__ unsigned int cnt;
    unsigned int lo = 0u, hi = 0xFFFFFFFFu;
    for (int it = 0; it < 32; ++it) {
        const unsigned int mid = lo + ((hi - lo) >> 1);
        if (tid == 0) cnt = 0u;
        __syncthreads();
        unsigned int c = 0;
#pragma unroll
        for (int j = 0; j < 8; ++j) c += (keys[j] <= mid) ? 1u : 0u;
        atomicAdd(&cnt, c);
        __syncthreads();
        const unsigned int total = cnt;
        __syncthreads();
        if (total >= 1024u) hi = mid; else lo = mid + 1u;
    }
    const unsigned int theta = lo;
#pragma unroll
    for (int j = 0; j < 8; ++j)
        mrow[tid + j * 256] = (keys[j] <= theta) ? 1 : 0;
}

// out_t = (mask_t & ~mask_f) ? 0.5*(x_t+x_f) : x_t ; symmetric for out_f
__global__ void combine_kernel(const float* __restrict__ x_t, const float* __restrict__ x_f,
                               const unsigned char* __restrict__ masks, float* __restrict__ out) {
    const int i = blockIdx.x * blockDim.x + threadIdx.x;
    if (i >= NLD) return;
    const int rl = i >> 7; // (n*L + l)
    const bool mt = masks[rl] != 0;
    const bool mf = masks[NLROWS + rl] != 0;
    const float xt = x_t[i];
    const float xf = x_f[i];
    const float avg = 0.5f * (xt + xf);
    out[i]       = (mt && !mf) ? avg : xt;
    out[NLD + i] = (mf && !mt) ? avg : xf;
}

extern "C" void kernel_launch(void* const* d_in, const int* in_sizes, int n_in,
                              void* d_out, int out_size, void* d_ws, size_t ws_size,
                              hipStream_t stream) {
    const float* x_t = (const float*)d_in[0];
    const float* x_f = (const float*)d_in[1];
    const float* W   = (const float*)d_in[2];
    const float* b   = (const float*)d_in[3];
    float* out = (float*)d_out;

    // workspace layout
    unsigned short* pt = (unsigned short*)d_ws;            // NLD bf16 = 4 MB
    unsigned short* pf = pt + NLD;                         // 4 MB
    unsigned short* Wb = pf + NLD;                         // 32 KB
    float* score = (float*)(Wb + DD * DD);                 // 2*N*L f32 = 128 KB
    unsigned char* masks = (unsigned char*)(score + 2 * NLROWS); // 32 KB

    // 1) zero score accumulators (atomics + graph replay => must re-zero each call)
    zero_f32_kernel<<<(2 * NLROWS + 255) / 256, 256, 0, stream>>>(score, 2 * NLROWS);

    // 2) W -> bf16
    convert_w_kernel<<<(DD * DD + 255) / 256, 256, 0, stream>>>(W, Wb);

    // 3) projections (WMMA bf16)
    proj_kernel<<<dim3(NLROWS / 16, 2), 128, 0, stream>>>(x_t, x_f, Wb, b, pt, pf);

    // 4) column scores, two-pass softmax (WMMA bf16) — only score_t/score_f are live
    score_kernel<<<dim3(LL / 32, NB, 2), 128, 0, stream>>>(pt, pf, score);

    // 5) k-th order statistic -> masks
    mask_kernel<<<16, 256, 0, stream>>>(score, masks);

    // 6) combine
    combine_kernel<<<(NLD + 255) / 256, 256, 0, stream>>>(x_t, x_f, masks, out);
}
